// Net_86466281603134
// MI455X (gfx1250) — compile-verified, matrix operand order checked by reading, not run on hardware
//
#include <hip/hip_runtime.h>
#include <hip/hip_bf16.h>

typedef __attribute__((ext_vector_type(16))) _Float16 v16h;
typedef __attribute__((ext_vector_type(8)))  _Float16 v8h;
typedef __attribute__((ext_vector_type(8)))  float    v8f;

#define BATCH    8
#define CLEN     512
#define DMODEL   256
#define DINNER   512
#define DSTATE   16
#define DTRANK   16
#define NTOK     (BATCH * CLEN)     // 4096
#define INDIM    25
#define OUTDIM   12
#define FLEN     128

// ---------------------------------------------------------------------------
// f16 fragment loader (v_wmma_f32_16x16x32_f16 ISA layout): a lane's 16 halfs
// are K = {khalf+0..7, 16+khalf+0..7}; with khalf folded into the pointer this
// is two contiguous 16-byte loads -> 2 x global_load_b128, zero converts.
// ---------------------------------------------------------------------------
__device__ __forceinline__ v16h load_frag_f16(const _Float16* __restrict__ p)
{
    const v8h lo = *(const v8h*)(p);
    const v8h hi = *(const v8h*)(p + 16);
    v16h f;
#pragma unroll
    for (int e = 0; e < 8; ++e) { f[e] = lo[e]; f[8 + e] = hi[e]; }
    return f;
}

__device__ __forceinline__ float apply_act(float v, int act)
{
    if (act == 1)      v = tanhf(v);
    else if (act == 2) v = (v > 20.f) ? v : log1pf(expf(v));
    return v;
}

// ---------------------------------------------------------------------------
// FAST GEMM (f16 operands, f32 accumulate): C = act( A @ W^T + bias ) [+resid]
// Requirements: M%64==0, N%16==0, K%32==0; A,W dense f16 (unit K stride).
// One wave -> 64x16 C supertile: 1 B fragment reused across 4 A fragments.
// Inner loop = 10 x b128 loads + 4 x v_wmma per K-step.
// ---------------------------------------------------------------------------
__global__ void wmma_gemm_fast_kernel(const _Float16* __restrict__ A,
                                      const _Float16* __restrict__ W,
                                      const float* __restrict__ bias,
                                      const float* __restrict__ resid,
                                      float* __restrict__ C,
                                      int M, int N, int K,
                                      long rsA, long rsW,
                                      long rsC, long csC,
                                      int act)
{
    const int lane   = threadIdx.x & 31;
    const int waveId = threadIdx.x >> 5;
    const int tilesN = N >> 4;
    const int tile   = blockIdx.x * 8 + waveId;          // wave-uniform
    if (tile >= (M >> 6) * tilesN) return;               // uniform exit

    const int tm    = (tile / tilesN) << 6;              // 64-row supertile
    const int tn    = (tile % tilesN) << 4;
    const int l15   = lane & 15;
    const int khalf = (lane >> 4) << 3;                  // 0 or 8

    const _Float16* pW  = W + (long)(tn + l15) * rsW + khalf;
    const _Float16* pA0 = A + (long)(tm +  0 + l15) * rsA + khalf;
    const _Float16* pA1 = A + (long)(tm + 16 + l15) * rsA + khalf;
    const _Float16* pA2 = A + (long)(tm + 32 + l15) * rsA + khalf;
    const _Float16* pA3 = A + (long)(tm + 48 + l15) * rsA + khalf;

    v8f acc0 = {}, acc1 = {}, acc2 = {}, acc3 = {};

    for (int k0 = 0; k0 < K; k0 += 32) {
        __builtin_prefetch((const void*)(pW + k0 + 32), 0, 0);
        const v16h bf = load_frag_f16(pW  + k0);
        const v16h a0 = load_frag_f16(pA0 + k0);
        const v16h a1 = load_frag_f16(pA1 + k0);
        const v16h a2 = load_frag_f16(pA2 + k0);
        const v16h a3 = load_frag_f16(pA3 + k0);
        acc0 = __builtin_amdgcn_wmma_f32_16x16x32_f16(false, a0, false, bf, (short)0, acc0, false, false);
        acc1 = __builtin_amdgcn_wmma_f32_16x16x32_f16(false, a1, false, bf, (short)0, acc1, false, false);
        acc2 = __builtin_amdgcn_wmma_f32_16x16x32_f16(false, a2, false, bf, (short)0, acc2, false, false);
        acc3 = __builtin_amdgcn_wmma_f32_16x16x32_f16(false, a3, false, bf, (short)0, acc3, false, false);
    }

    const int   col  = tn + l15;
    const float bcol = bias ? bias[col] : 0.f;
    const long  cOff = (long)col * csC;

#pragma unroll
    for (int s = 0; s < 4; ++s) {
        const v8f acc = (s == 0) ? acc0 : (s == 1) ? acc1 : (s == 2) ? acc2 : acc3;
        const int rbase = tm + (s << 4) + khalf;
#pragma unroll
        for (int r = 0; r < 8; ++r) {
            const long ci = (long)(rbase + r) * rsC + cOff;
            float v = acc[r] + bcol;
            if (resid) v += resid[ci];
            C[ci] = apply_act(v, act);
        }
    }
}

// ---------------------------------------------------------------------------
// GENERAL GEMM (f32 operands, arbitrary shapes/strides/batch). Branch-free
// loads via clamped indices + arithmetic zeroing; guard-free main K body plus
// one guarded tail block.  act: 0 = none, 1 = tanh, 2 = softplus
// ---------------------------------------------------------------------------
__global__ void wmma_gemm_gen_kernel(const float* __restrict__ A,
                                     const float* __restrict__ W,
                                     const float* __restrict__ bias,
                                     const float* __restrict__ resid,
                                     float* __restrict__ C,
                                     int M, int N, int K,
                                     long rsA, long csA,
                                     long rsW, long csW,
                                     long rsC, long csC,
                                     long bsA, long bsC,
                                     int act)
{
    A += (long)blockIdx.y * bsA;
    C += (long)blockIdx.y * bsC;
    if (resid) resid += (long)blockIdx.y * bsC;

    const int lane   = threadIdx.x & 31;
    const int waveId = threadIdx.x >> 5;
    const int tilesN = (N + 15) >> 4;
    const int tilesM = (M + 15) >> 4;
    const int tile   = blockIdx.x * 8 + waveId;          // wave-uniform
    if (tile >= tilesM * tilesN) return;                 // uniform exit

    const int tm    = (tile / tilesN) << 4;
    const int tn    = (tile % tilesN) << 4;
    const int l15   = lane & 15;
    const int khalf = (lane >> 4) << 3;                  // 0 or 8

    const int mrow = tm + l15;
    const int ncol = tn + l15;
    const float mval = (mrow < M) ? 1.f : 0.f;
    const float nval = (ncol < N) ? 1.f : 0.f;
    const float* pA = A + (long)((mrow < M) ? mrow : (M - 1)) * rsA;
    const float* pW = W + (long)((ncol < N) ? ncol : (N - 1)) * rsW;

    long ofA[16], ofW[16];
#pragma unroll
    for (int e = 0; e < 16; ++e) {
        const int ke = e + ((e >= 8) ? 8 : 0) + khalf;   // ISA 16-bit frag K index
        ofA[e] = (long)ke * csA;
        ofW[e] = (long)ke * csW;
    }

    v8f acc = {};
    const int Kfull = K & ~31;
    long baseA = 0, baseW = 0;
    for (int k0 = 0; k0 < Kfull; k0 += 32) {
        v16h af, bf;
#pragma unroll
        for (int e = 0; e < 16; ++e) {
            af[e] = (_Float16)(mval * pA[baseA + ofA[e]]);
            bf[e] = (_Float16)(nval * pW[baseW + ofW[e]]);
        }
        acc = __builtin_amdgcn_wmma_f32_16x16x32_f16(false, af, false, bf, (short)0, acc, false, false);
        baseA += 32 * csA;
        baseW += 32 * csW;
    }
    if (Kfull < K) {                                     // single guarded tail block
        v16h af, bf;
#pragma unroll
        for (int e = 0; e < 16; ++e) {
            const int ke = Kfull + e + ((e >= 8) ? 8 : 0) + khalf;
            const int kc = (ke < K) ? ke : (K - 1);      // clamp -> load always legal
            const float a = mval * pA[(long)kc * csA];
            const float b = nval * pW[(long)kc * csW];
            af[e] = (_Float16)((ke < K) ? a : 0.f);
            bf[e] = (_Float16)((ke < K) ? b : 0.f);
        }
        acc = __builtin_amdgcn_wmma_f32_16x16x32_f16(false, af, false, bf, (short)0, acc, false, false);
    }

    const int col = tn + l15;
#pragma unroll
    for (int r = 0; r < 8; ++r) {
        const int row = tm + r + khalf;
        if (row < M && col < N) {
            float v = acc[r];
            if (bias)  v += bias[col];
            if (resid) v += resid[(long)row * rsC + (long)col * csC];
            C[(long)row * rsC + (long)col * csC] = apply_act(v, act);
        }
    }
}

// ---------------------------------------------------------------------------
// f32 -> f16 weight conversion (once per launch; weights reused 4096x).
// ---------------------------------------------------------------------------
__global__ void f32_to_f16_kernel(const float* __restrict__ src,
                                  _Float16* __restrict__ dst, long n)
{
    const long i = (long)blockIdx.x * blockDim.x + threadIdx.x;
    if (i < n) dst[i] = (_Float16)src[i];
}

// ---------------------------------------------------------------------------
// Build x = concat([x_time[...,None], x_mask, x_vals], -1)  -> (NTOK, 25)
// ---------------------------------------------------------------------------
__global__ void build_x_kernel(const float* __restrict__ x_time,
                               const float* __restrict__ x_vals,
                               const float* __restrict__ x_mask,
                               float* __restrict__ xbuf)
{
    const long idx = (long)blockIdx.x * blockDim.x + threadIdx.x;
    if (idx >= (long)NTOK * INDIM) return;
    const long t = idx / INDIM;
    const int  c = (int)(idx % INDIM);
    float v;
    if (c == 0)       v = x_time[t];
    else if (c <= 12) v = x_mask[t * OUTDIM + (c - 1)];
    else              v = x_vals[t * OUTDIM + (c - 13)];
    xbuf[idx] = v;
}

// ---------------------------------------------------------------------------
// RMSNorm over last dim (D = 256).  One block (256 threads) per token.
// Emits both f32 (general GEMM consumer) and f16 (fast GEMM operand).
// ---------------------------------------------------------------------------
__global__ void rmsnorm_kernel(const float* __restrict__ x,
                               const float* __restrict__ w,
                               float* __restrict__ out32,
                               _Float16* __restrict__ out16)
{
    __shared__ float red[8];
    const long t = blockIdx.x;
    const float v = x[t * DMODEL + threadIdx.x];
    float s = v * v;
#pragma unroll
    for (int off = 16; off > 0; off >>= 1) s += __shfl_down(s, off, 32);
    if ((threadIdx.x & 31) == 0) red[threadIdx.x >> 5] = s;
    __syncthreads();
    if (threadIdx.x == 0) {
        float tot = 0.f;
#pragma unroll
        for (int i = 0; i < 8; ++i) tot += red[i];
        red[0] = rsqrtf(tot / (float)DMODEL + 1e-5f);
    }
    __syncthreads();
    const float o = v * red[0] * w[threadIdx.x];
    out32[t * DMODEL + threadIdx.x] = o;
    out16[t * DMODEL + threadIdx.x] = (_Float16)o;
}

// ---------------------------------------------------------------------------
// Depthwise causal conv (width 4) over sequence + SiLU.
// Emits f32 (scan consumer) and f16 (x_proj fast GEMM operand).
// ---------------------------------------------------------------------------
__global__ void conv_silu_kernel(const float* __restrict__ xz,
                                 const float* __restrict__ cw,
                                 const float* __restrict__ cb,
                                 float* __restrict__ xb2,
                                 _Float16* __restrict__ xb2h)
{
    const long idx = (long)blockIdx.x * blockDim.x + threadIdx.x;
    if (idx >= (long)NTOK * DINNER) return;
    const int  d = (int)(idx & (DINNER - 1));
    const long t = idx >> 9;
    const int  l = (int)(t & (CLEN - 1));
    float acc = cb[d];
#pragma unroll
    for (int j = 0; j < 4; ++j) {
        const int ls = l - 3 + j;
        if (ls >= 0) acc += cw[d * 4 + j] * xz[(t - 3 + j) * 1024L + d];
    }
    const float o = acc / (1.f + expf(-acc));            // SiLU
    xb2[idx]  = o;
    xb2h[idx] = (_Float16)o;
}

// ---------------------------------------------------------------------------
// Selective scan. Grid = BATCH blocks, 512 threads (one per channel d).
// Output f16 only (sole consumer is the out_proj fast GEMM).
// ---------------------------------------------------------------------------
__global__ void scan_kernel(const float* __restrict__ dbc,    // (NTOK, 48)
                            const float* __restrict__ delta,  // (NTOK, 512)
                            const float* __restrict__ xb2,    // (NTOK, 512)
                            const float* __restrict__ xz,     // z at cols 512..1023
                            const float* __restrict__ A_log,  // (512, 16)
                            const float* __restrict__ Dp,     // (512,)
                            _Float16* __restrict__ ysh)       // (NTOK, 512) f16
{
    __shared__ float sBC[2 * DSTATE];
    const int b = blockIdx.x;
    const int d = threadIdx.x;

    float a[DSTATE], st[DSTATE];
#pragma unroll
    for (int n = 0; n < DSTATE; ++n) { a[n] = -expf(A_log[d * DSTATE + n]); st[n] = 0.f; }
    const float Dd = Dp[d];

    for (int l = 0; l < CLEN; ++l) {
        const long t = (long)b * CLEN + l;
        if (threadIdx.x < 2 * DSTATE)
            sBC[threadIdx.x] = dbc[t * 48 + DTRANK + threadIdx.x];
        __syncthreads();

        const float dl = delta[t * DINNER + d];
        const float xv = xb2[t * DINNER + d];
        const float zv = xz[t * 1024L + DINNER + d];
        const float dx = dl * xv;
        float y = 0.f;
#pragma unroll
        for (int n = 0; n < DSTATE; ++n) {
            st[n] = expf(dl * a[n]) * st[n] + dx * sBC[n];
            y += st[n] * sBC[DSTATE + n];
        }
        y += Dd * xv;
        ysh[t * DINNER + d] = (_Float16)(y * (zv / (1.f + expf(-zv))));
        __syncthreads();
    }
}

// ---------------------------------------------------------------------------
// Tuple tail: copy y_vals and bool(y_mask) into output slots 2 and 3.
// ---------------------------------------------------------------------------
__global__ void tail_copy_kernel(const float* __restrict__ yv,
                                 const float* __restrict__ ym,
                                 float* __restrict__ out)
{
    const int i = blockIdx.x * blockDim.x + threadIdx.x;
    if (i < BATCH * FLEN * OUTDIM) {
        out[BATCH * FLEN * OUTDIM + i]     = yv[i];
        out[2 * BATCH * FLEN * OUTDIM + i] = (ym[i] != 0.f) ? 1.f : 0.f;
    }
}

// ---------------------------------------------------------------------------
static inline void launch_fast(hipStream_t s,
                               const _Float16* A, const _Float16* W,
                               const float* bias, const float* resid, float* C,
                               int M, int N, int K, long rsA, long rsW,
                               long rsC, long csC, int act)
{
    const int tiles = (M >> 6) * (N >> 4);
    wmma_gemm_fast_kernel<<<dim3((tiles + 7) / 8), 256, 0, s>>>(
        A, W, bias, resid, C, M, N, K, rsA, rsW, rsC, csC, act);
}

static inline void launch_gen(hipStream_t s,
                              const float* A, const float* W, const float* bias,
                              const float* resid, float* C,
                              int M, int N, int K,
                              long rsA, long csA, long rsW, long csW,
                              long rsC, long csC,
                              long bsA, long bsC, int batches, int act)
{
    const int tiles = ((M + 15) / 16) * ((N + 15) / 16);
    wmma_gemm_gen_kernel<<<dim3((tiles + 7) / 8, batches), 256, 0, s>>>(
        A, W, bias, resid, C, M, N, K, rsA, csA, rsW, csW, rsC, csC, bsA, bsC, act);
}

static inline void cvt_f16(hipStream_t s, const float* src, _Float16* dst, long n)
{
    f32_to_f16_kernel<<<(unsigned)((n + 255) / 256), 256, 0, s>>>(src, dst, n);
}

extern "C" void kernel_launch(void* const* d_in, const int* in_sizes, int n_in,
                              void* d_out, int out_size, void* d_ws, size_t ws_size,
                              hipStream_t stream)
{
    // ---- inputs (setup_inputs() dict insertion order, params nested last) ----
    const float* x_time = (const float*)d_in[0];
    const float* x_vals = (const float*)d_in[1];
    const float* x_mask = (const float*)d_in[2];
    const float* y_vals = (const float*)d_in[4];
    const float* y_mask = (const float*)d_in[5];

    // per-layer params: in_proj, conv_w, conv_b, x_proj, dt_w, dt_b, A_log, D, out_proj, norm_w
    const float* L[4][10];
    for (int i = 0; i < 4; ++i)
        for (int j = 0; j < 10; ++j)
            L[i][j] = (const float*)d_in[8 + i * 10 + j];
    const float* norm_f    = (const float*)d_in[48];
    const float* lin_in_w  = (const float*)d_in[49];
    const float* lin_in_b  = (const float*)d_in[50];
    const float* lin_out_w = (const float*)d_in[51];
    const float* lin_out_b = (const float*)d_in[52];
    const float* time_w    = (const float*)d_in[53];
    const float* time_b    = (const float*)d_in[54];

    // ---- workspace carve-up ----
    float* ws = (float*)d_ws;
    size_t off = 0;
    float* xbuf  = ws + off; off += (size_t)NTOK * INDIM;
    float* h     = ws + off; off += (size_t)NTOK * DMODEL;
    float* hn    = ws + off; off += (size_t)NTOK * DMODEL;
    float* xzbuf = ws + off; off += (size_t)NTOK * 1024;
    float* xb2   = ws + off; off += (size_t)NTOK * DINNER;
    float* dbc   = ws + off; off += (size_t)NTOK * 48;
    float* delta = ws + off; off += (size_t)NTOK * DINNER;
    float* obuf  = ws + off; off += (size_t)NTOK * OUTDIM;

    _Float16* hws = (_Float16*)(ws + off);
    size_t hoff = 0;
    _Float16* hnh   = hws + hoff; hoff += (size_t)NTOK * DMODEL;
    _Float16* xb2h  = hws + hoff; hoff += (size_t)NTOK * DINNER;
    _Float16* ysh   = hws + hoff; hoff += (size_t)NTOK * DINNER;
    _Float16* ipw_h[4], * xpw_h[4], * opw_h[4];
    for (int i = 0; i < 4; ++i) {
        ipw_h[i] = hws + hoff; hoff += (size_t)1024 * DMODEL;    // in_proj  (1024,256)
        xpw_h[i] = hws + hoff; hoff += (size_t)48 * DINNER;      // x_proj   (48,512)
        opw_h[i] = hws + hoff; hoff += (size_t)DMODEL * DINNER;  // out_proj (256,512)
    }

    float* out = (float*)d_out;

    // 0) convert fast-GEMM weights to f16 (reused 4096x each)
    for (int i = 0; i < 4; ++i) {
        cvt_f16(stream, L[i][0], ipw_h[i], (long)1024 * DMODEL);
        cvt_f16(stream, L[i][3], xpw_h[i], (long)48 * DINNER);
        cvt_f16(stream, L[i][8], opw_h[i], (long)DMODEL * DINNER);
    }

    // 1) input concat
    {
        const long n = (long)NTOK * INDIM;
        build_x_kernel<<<(unsigned)((n + 255) / 256), 256, 0, stream>>>(x_time, x_vals, x_mask, xbuf);
    }
    // 2) lin_in: h = x @ lin_in_w^T + b        (4096x25 @ 25x256)  [general: K=25]
    launch_gen(stream, xbuf, lin_in_w, lin_in_b, nullptr, h,
               NTOK, DMODEL, INDIM, INDIM, 1, INDIM, 1, DMODEL, 1, 0, 0, 1, 0);

    // 3) mamba layers
    for (int i = 0; i < 4; ++i) {
        const float* conv_w = L[i][1];
        const float* conv_b = L[i][2];
        const float* dt_w   = L[i][4];
        const float* dt_b   = L[i][5];
        const float* A_log  = L[i][6];
        const float* Dp     = L[i][7];
        const float* norm_w = L[i][9];

        rmsnorm_kernel<<<NTOK, DMODEL, 0, stream>>>(h, norm_w, hn, hnh);

        // xz = hn @ in_proj^T                  (4096x256 @ 256x1024)  [fast f16]
        launch_fast(stream, hnh, ipw_h[i], nullptr, nullptr, xzbuf,
                    NTOK, 1024, DMODEL, DMODEL, DMODEL, 1024, 1, 0);

        // depthwise causal conv + SiLU on xb half (dual f32/f16 output)
        {
            const long n = (long)NTOK * DINNER;
            conv_silu_kernel<<<(unsigned)((n + 255) / 256), 256, 0, stream>>>(
                xzbuf, conv_w, conv_b, xb2, xb2h);
        }

        // dbc = xb2 @ x_proj^T                 (4096x512 @ 512x48)  [fast f16]
        launch_fast(stream, xb2h, xpw_h[i], nullptr, nullptr, dbc,
                    NTOK, 48, DINNER, DINNER, DINNER, 48, 1, 0);

        // delta = softplus(dbc[:, :16] @ dt_w^T + dt_b)  (4096x16 @ 16x512) [general]
        launch_gen(stream, dbc, dt_w, dt_b, nullptr, delta,
                   NTOK, DINNER, DTRANK, 48, 1, DTRANK, 1, DINNER, 1, 0, 0, 1, 2);

        // selective scan -> f16 ys
        scan_kernel<<<BATCH, DINNER, 0, stream>>>(dbc, delta, xb2, xzbuf, A_log, Dp, ysh);

        // h = h + ys @ out_proj^T              (4096x512 @ 512x256)  [fast f16]
        launch_fast(stream, ysh, opw_h[i], nullptr, h, h,
                    NTOK, DMODEL, DINNER, DINNER, DINNER, DMODEL, 1, 0);
    }

    // 4) final rmsnorm (f32 side feeds lin_out general GEMM)
    rmsnorm_kernel<<<NTOK, DMODEL, 0, stream>>>(h, norm_f, hn, hnh);

    // 5) o = tanh(hn @ lin_out_w^T + b)        (4096x256 @ 256x12)  [general: N=12]
    launch_gen(stream, hn, lin_out_w, lin_out_b, nullptr, obuf,
               NTOK, OUTDIM, DMODEL, DMODEL, 1, DMODEL, 1, OUTDIM, 1, 0, 0, 1, 1);

    // 6) y_hat[b] = o[b]^T @ time_w^T + time_b   batched (12x512 @ 512x128) [general]
    launch_gen(stream, obuf, time_w, time_b, nullptr, out,
               OUTDIM, FLEN, CLEN,
               1, OUTDIM, CLEN, 1, 1, OUTDIM,
               (long)CLEN * OUTDIM, (long)FLEN * OUTDIM, BATCH, 0);

    // 7) tuple tail: y_vals flat, bool(y_mask)
    {
        const int n = BATCH * FLEN * OUTDIM;
        tail_copy_kernel<<<(n + 255) / 256, 256, 0, stream>>>(y_vals, y_mask, out);
    }
    (void)in_sizes; (void)n_in; (void)out_size; (void)ws_size;
}
